// Memoryx_77558519432022
// MI455X (gfx1250) — matrix operand order repaired and checked by reading, not compiled, not run on hardware
//
#include <hip/hip_runtime.h>
#include <hip/hip_bf16.h>

typedef __attribute__((ext_vector_type(16))) _Float16 v16h;
typedef __attribute__((ext_vector_type(8)))  _Float16 v8h;
typedef __attribute__((ext_vector_type(8)))  float    v8f;
typedef __attribute__((ext_vector_type(4)))  float    v4f;

#define NUM_ITEM 2048
#define C_DIM    512
#define ROWS_WG  128
#define WAVES    (ROWS_WG / 16)
#define THREADS  (WAVES * 32)
#define KQ       8
#define LO_SCALE 2048.0f
#define LO_INV   (1.0f / 2048.0f)

union V16 { v16h v; v8h g[2]; };

// ---------------------------------------------------------------------------
// Phase 0: split mempool f32 -> f16 hi plane + (scaled) f16 lo plane in d_ws.
// Row-major [item][c] so each WMMA A-operand is two contiguous 16B loads.
// ---------------------------------------------------------------------------
__global__ __launch_bounds__(256) void prep_kernel(const float* __restrict__ mp,
                                                   _Float16* __restrict__ hi,
                                                   _Float16* __restrict__ lo) {
    int i = blockIdx.x * 256 + threadIdx.x;
    if (i < NUM_ITEM * C_DIM) {
        float v = mp[i];
        _Float16 h = (_Float16)v;
        hi[i] = h;
        lo[i] = (_Float16)((v - (float)h) * LO_SCALE);
    }
}

// ---------------------------------------------------------------------------
// Fused: 128-row q tile x all 2048 items via f16 hi/lo WMMA (M=32 per step so
// each LDS B load feeds two A tiles), lane-private top-8, softmax, combine.
// ---------------------------------------------------------------------------
__global__ __launch_bounds__(THREADS) void moe_topk_kernel(
        const float* __restrict__ x,
        const float* __restrict__ mp,
        const _Float16* __restrict__ wsHi,
        const _Float16* __restrict__ wsLo,
        float* __restrict__ out) {
    // B operands pre-shuffled: [wave][kb][lane][slot] halves (hi and lo planes)
    __shared__ __align__(16) _Float16 ldsBhi[WAVES * 16 * 32 * 16];
    __shared__ __align__(16) _Float16 ldsBlo[WAVES * 16 * 32 * 16];

    const int tid  = threadIdx.x;
    const int lane = tid & 31;
    const int wv   = tid >> 5;
    const int n0   = blockIdx.x * ROWS_WG;       // first row of tile
    const int b    = n0 >> 10;                   // batch (1024 rows per batch)
    const int s0   = n0 & 1023;                  // spatial offset h*32+w
    const long xBase = (long)b * (C_DIM * 1024) + s0;

    // ---- load q tile: coalesced per-channel, split hi/lo, scatter to
    // ---- B-operand layout in this code's contiguous k order (lane<16: K%32 in 0..15, lane>=16: 16..31)
    for (int e = tid; e < ROWS_WG * C_DIM; e += THREADS) {
        int row = e & (ROWS_WG - 1);
        int c   = e >> 7;                        // log2(ROWS_WG)
        float v = x[xBase + (long)c * 1024 + row];
        _Float16 h = (_Float16)v;
        _Float16 l = (_Float16)((v - (float)h) * LO_SCALE);
        int w2 = row >> 4, n16 = row & 15;
        int kb = c >> 5,  j   = c & 31;
        int bl   = n16 + (j & 16);               // lane within wave
        int slot = j & 15;
        int idx = (((w2 * 16 + kb) * 32) + bl) * 16 + slot;
        ldsBhi[idx] = h;
        ldsBlo[idx] = l;
    }
    __syncthreads();

    // ---- running top-8 (sorted descending), lane-private
    float tv[KQ];
    int   ti[KQ];
#pragma unroll
    for (int j = 0; j < KQ; ++j) { tv[j] = -3.4e38f; ti[j] = 0; }

    const int itemOfs = (lane & 16) ? 8 : 0;     // D-layout M offset per half-wave
    const int aKOfs   = (lane & 16) ? 16 : 0;    // contiguous A-layout K offset
    const _Float16* ldsbh = ldsBhi + wv * 8192 + lane * 16;
    const _Float16* ldsbl = ldsBlo + wv * 8192 + lane * 16;

    for (int it = 0; it < NUM_ITEM / 32; ++it) {
        const int itm0 = it * 32 + (lane & 15);      // chunk 0 item row
        const _Float16* aH0 = wsHi + itm0 * C_DIM + aKOfs;
        const _Float16* aL0 = wsLo + itm0 * C_DIM + aKOfs;
        const _Float16* aH1 = aH0 + 16 * C_DIM;      // chunk 1 item row
        const _Float16* aL1 = aL0 + 16 * C_DIM;
        v8f accH0 = {}, accL0 = {};
        v8f accH1 = {}, accL1 = {};
#pragma unroll 4
        for (int kb = 0; kb < 16; ++kb) {
            V16 bhi, blo;
            bhi.g[0] = *(const v8h*)(ldsbh + kb * 512);
            bhi.g[1] = *(const v8h*)(ldsbh + kb * 512 + 8);
            blo.g[0] = *(const v8h*)(ldsbl + kb * 512);
            blo.g[1] = *(const v8h*)(ldsbl + kb * 512 + 8);
            V16 ahi, alo;
            ahi.g[0] = *(const v8h*)(aH0 + kb * 32);       // K = kb*32+ofs+0..7
            ahi.g[1] = *(const v8h*)(aH0 + kb * 32 + 8);   // K = kb*32+ofs+8..15
            alo.g[0] = *(const v8h*)(aL0 + kb * 32);
            alo.g[1] = *(const v8h*)(aL0 + kb * 32 + 8);
            accH0 = __builtin_amdgcn_wmma_f32_16x16x32_f16(
                        false, ahi.v, false, bhi.v, (short)0, accH0, false, false);
            accL0 = __builtin_amdgcn_wmma_f32_16x16x32_f16(
                        false, ahi.v, false, blo.v, (short)0, accL0, false, false);
            accL0 = __builtin_amdgcn_wmma_f32_16x16x32_f16(
                        false, alo.v, false, bhi.v, (short)0, accL0, false, false);
            ahi.g[0] = *(const v8h*)(aH1 + kb * 32);
            ahi.g[1] = *(const v8h*)(aH1 + kb * 32 + 8);
            alo.g[0] = *(const v8h*)(aL1 + kb * 32);
            alo.g[1] = *(const v8h*)(aL1 + kb * 32 + 8);
            accH1 = __builtin_amdgcn_wmma_f32_16x16x32_f16(
                        false, ahi.v, false, bhi.v, (short)0, accH1, false, false);
            accL1 = __builtin_amdgcn_wmma_f32_16x16x32_f16(
                        false, ahi.v, false, blo.v, (short)0, accL1, false, false);
            accL1 = __builtin_amdgcn_wmma_f32_16x16x32_f16(
                        false, alo.v, false, bhi.v, (short)0, accL1, false, false);
        }
        // compensated scores for this lane's row: 16 items (2 chunks x 8)
        float s0[8], s1[8];
        float cmax = -3.4e38f;
#pragma unroll
        for (int v = 0; v < 8; ++v) {
            s0[v] = accH0[v] + accL0[v] * LO_INV;
            s1[v] = accH1[v] + accL1[v] * LO_INV;
            cmax = fmaxf(cmax, fmaxf(s0[v], s1[v]));
        }
        // wave-uniform skip: nothing here can enter any lane's top-8
        if (__ballot(cmax > tv[KQ - 1]) != 0ull) {
#pragma unroll
            for (int v = 0; v < 16; ++v) {
                float cv = (v < 8) ? s0[v] : s1[v - 8];
                int   ci = it * 32 + (v < 8 ? 0 : 8) + (v & 7) + itemOfs;
#pragma unroll
                for (int j = 0; j < KQ; ++j) {
                    bool  gt = cv > tv[j];
                    float nv = gt ? cv : tv[j];
                    int   ni = gt ? ci : ti[j];
                    float dv = gt ? tv[j] : cv;
                    int   di = gt ? ti[j] : ci;
                    tv[j] = nv; ti[j] = ni; cv = dv; ci = di;
                }
            }
        }
    }

    // ---- merge with partner lane (same row, other item half)
    float ov[KQ]; int oi[KQ];
#pragma unroll
    for (int j = 0; j < KQ; ++j) {
        ov[j] = __shfl_xor(tv[j], 16, 32);
        oi[j] = __shfl_xor(ti[j], 16, 32);
    }
#pragma unroll
    for (int v = 0; v < KQ; ++v) {
        float cv = ov[v]; int ci = oi[v];
#pragma unroll
        for (int j = 0; j < KQ; ++j) {
            bool  gt = cv > tv[j];
            float nv = gt ? cv : tv[j];
            int   ni = gt ? ci : ti[j];
            float dv = gt ? tv[j] : cv;
            int   di = gt ? ti[j] : ci;
            tv[j] = nv; ti[j] = ni; cv = dv; ci = di;
        }
    }

    // ---- softmax over top-8 (list is sorted desc -> tv[0] is max)
    float m0 = tv[0];
    float sum = 0.f;
#pragma unroll
    for (int j = 0; j < KQ; ++j) { tv[j] = __expf(tv[j] - m0); sum += tv[j]; }
    float inv = 1.0f / sum;
#pragma unroll
    for (int j = 0; j < KQ; ++j) tv[j] *= inv;

    // ---- combine: out[n, c] = sum_j w_j * mempool[idx_j, c], written back in
    // ---- B,C,H,W order. lanes 0..15: c in [0,256), lanes 16..31: [256,512);
    // ---- per store, 16 consecutive rows -> full 64B lines. float4 gathers.
    const int row = lane & 15;
    const int ch  = lane >> 4;
    const float* mrow[KQ];
#pragma unroll
    for (int j = 0; j < KQ; ++j) mrow[j] = mp + ti[j] * C_DIM + ch * 256;
    const long oBase = xBase + wv * 16 + row + (long)(ch * 256) * 1024;
    for (int c4 = 0; c4 < 256 / 4; ++c4) {
        v4f a = {};
#pragma unroll
        for (int j = 0; j < KQ; ++j) {
            v4f m = *(const v4f*)(mrow[j] + c4 * 4);
            a += tv[j] * m;
        }
#pragma unroll
        for (int i = 0; i < 4; ++i)
            out[oBase + (long)(c4 * 4 + i) * 1024] = a[i];
    }
}

extern "C" void kernel_launch(void* const* d_in, const int* in_sizes, int n_in,
                              void* d_out, int out_size, void* d_ws, size_t ws_size,
                              hipStream_t stream) {
    (void)n_in; (void)out_size; (void)ws_size;
    const float* x  = (const float*)d_in[0];
    const float* mp = (const float*)d_in[1];
    // d_in[2] is k (== 8, fixed by the reference setup)
    float* out = (float*)d_out;

    _Float16* wsHi = (_Float16*)d_ws;
    _Float16* wsLo = wsHi + NUM_ITEM * C_DIM;

    prep_kernel<<<(NUM_ITEM * C_DIM + 255) / 256, 256, 0, stream>>>(mp, wsHi, wsLo);

    const int nRows = in_sizes[0] / C_DIM;   // 32768
    moe_topk_kernel<<<nRows / ROWS_WG, THREADS, 0, stream>>>(x, mp, wsHi, wsLo, out);
}
